// XianYuGatConv_65755949301970
// MI455X (gfx1250) — matrix operand order repaired and checked by reading, not compiled, not run on hardware
//
#include <hip/hip_runtime.h>

// ---------- CDNA5 WMMA types ----------
typedef __attribute__((ext_vector_type(16))) __bf16 v16bf;
typedef __attribute__((ext_vector_type(8)))  float  v8f;

// Wave index within block, forced into an SGPR so tile math stays scalar.
__device__ __forceinline__ int wave_in_block() {
  return __builtin_amdgcn_readfirstlane((int)(threadIdx.x >> 5));
}

// ---------- fragment loaders (layouts per CDNA5 ISA §7.12.2) ----------
// A (16x32 bf16): lane l holds row m = l%16; element j -> K = (j<8 ? j : j+8) + 8*(l/16)
__device__ __forceinline__ v16bf load_A_rowmajor(const float* X, int ld, int row,
                                                 int kt, int half) {
  const float* r = X + row * ld;
  v16bf a;
#pragma unroll
  for (int j = 0; j < 16; ++j) {
    int K = kt * 32 + (j < 8 ? j : j + 8) + 8 * half;
    a[j] = (__bf16)r[K];
  }
  return a;
}

// A row is concat(src_feats[src_idx[e]][0:64], edge_feats[e][0:32])  (96 wide)
__device__ __forceinline__ v16bf load_A_edge(const float* __restrict__ src_feats,
                                             const float* __restrict__ edge_feats,
                                             const int* __restrict__ src_idx,
                                             int e, int kt, int half) {
  const float* srow = src_feats + src_idx[e] * 64;
  const float* erow = edge_feats + e * 32;
  v16bf a;
#pragma unroll
  for (int j = 0; j < 16; ++j) {
    int K = kt * 32 + (j < 8 ? j : j + 8) + 8 * half;
    float v = (K < 64) ? srow[K] : erow[K - 64];
    a[j] = (__bf16)v;
  }
  return a;
}

// B (32x16 bf16) = W^T slice: B[k][n] = W[n][kt*32+k]; lane l holds col n = l%16,
// element j -> K = (j<8 ? j : j+8) + 8*(l/16)
__device__ __forceinline__ v16bf load_B_wT(const float* W, int ld, int n,
                                           int kt, int half) {
  const float* wrow = W + n * ld;
  v16bf b;
#pragma unroll
  for (int j = 0; j < 16; ++j) {
    int K = kt * 32 + (j < 8 ? j : j + 8) + 8 * half;
    b[j] = (__bf16)wrow[K];
  }
  return b;
}

__device__ __forceinline__ void atomicMaxF(float* addr, float val) {
  if (val >= 0.0f)
    atomicMax((int*)addr, __float_as_int(val));
  else
    atomicMin((unsigned int*)addr, (unsigned int)__float_as_int(val));
}

// ---------- kernel 0: init ----------
__global__ void k_init(float* agg, float* amax, float* denom, int aggN, int N) {
  int stride = gridDim.x * blockDim.x;
  int i0 = blockIdx.x * blockDim.x + threadIdx.x;
  for (int i = i0; i < aggN; i += stride) agg[i] = 0.0f;
  for (int i = i0; i < N; i += stride) {
    amax[i] = -__builtin_inff();
    denom[i] = 0.0f;
  }
}

// ---------- kernel 1: dst_proj = dst_feats @ W_dst^T (nTiles*16 x 64) ----------
__global__ void k_dstproj(const float* __restrict__ dst_feats,
                          const float* __restrict__ W_dst,
                          float* __restrict__ dst_proj, int nTiles) {
  int lane = threadIdx.x & 31;
  int half = lane >> 4, c = lane & 15;
  int wave = blockIdx.x * (blockDim.x >> 5) + wave_in_block();
  int nWaves = gridDim.x * (blockDim.x >> 5);

  v16bf B[4][2];
#pragma unroll
  for (int nt = 0; nt < 4; ++nt)
#pragma unroll
    for (int kt = 0; kt < 2; ++kt)
      B[nt][kt] = load_B_wT(W_dst, 64, nt * 16 + c, kt, half);

  for (int t = wave; t < nTiles; t += nWaves) {
    int base = t * 16;
    v16bf A[2];
#pragma unroll
    for (int kt = 0; kt < 2; ++kt)
      A[kt] = load_A_rowmajor(dst_feats, 64, base + c, kt, half);

#pragma unroll
    for (int nt = 0; nt < 4; ++nt) {
      v8f acc = {0.f, 0.f, 0.f, 0.f, 0.f, 0.f, 0.f, 0.f};
#pragma unroll
      for (int kt = 0; kt < 2; ++kt)
        acc = __builtin_amdgcn_wmma_f32_16x16x32_bf16(false, A[kt], false, B[nt][kt],
                                                      (short)0, acc, false, false);
      float* outp = dst_proj + (base + 8 * half) * 64 + nt * 16 + c;
#pragma unroll
      for (int r = 0; r < 8; ++r) outp[r * 64] = acc[r];
    }
  }
}

// ---------- kernel 2: per-edge h_ue, a = h.dot(dst_proj[dst]), segment max ----------
__global__ void k_edge_pass1(const float* __restrict__ src_feats,
                             const float* __restrict__ edge_feats,
                             const int* __restrict__ src_idx,
                             const int* __restrict__ dst_idx,
                             const float* __restrict__ W_src,
                             const float* __restrict__ dst_proj,
                             float* __restrict__ a_out, float* __restrict__ amax,
                             int nTiles) {
  int lane = threadIdx.x & 31;
  int half = lane >> 4, c = lane & 15;
  int wave = blockIdx.x * (blockDim.x >> 5) + wave_in_block();
  int nWaves = gridDim.x * (blockDim.x >> 5);

  v16bf B[4][3];  // hoisted: reused across all tiles of this wave
#pragma unroll
  for (int nt = 0; nt < 4; ++nt)
#pragma unroll
    for (int kt = 0; kt < 3; ++kt)
      B[nt][kt] = load_B_wT(W_src, 96, nt * 16 + c, kt, half);

  for (int t = wave; t < nTiles; t += nWaves) {
    int base = t * 16;
    v16bf A[3];
#pragma unroll
    for (int kt = 0; kt < 3; ++kt)
      A[kt] = load_A_edge(src_feats, edge_feats, src_idx, base + c, kt, half);

    v8f acc[4];
#pragma unroll
    for (int nt = 0; nt < 4; ++nt) {
      acc[nt] = (v8f){0.f, 0.f, 0.f, 0.f, 0.f, 0.f, 0.f, 0.f};
#pragma unroll
      for (int kt = 0; kt < 3; ++kt)
        acc[nt] = __builtin_amdgcn_wmma_f32_16x16x32_bf16(false, A[kt], false, B[nt][kt],
                                                          (short)0, acc[nt], false, false);
    }

    // per-row dot with dst_proj[dst_idx[e]] : row M = r + 8*half, col = nt*16 + c
    float partial[8];
#pragma unroll
    for (int r = 0; r < 8; ++r) {
      const float* p = dst_proj + dst_idx[base + r + 8 * half] * 64 + c;
      partial[r] = acc[0][r] * p[0] + acc[1][r] * p[16] +
                   acc[2][r] * p[32] + acc[3][r] * p[48];
    }
    // reduce across the 16 lanes of each half (xor masks 1..8 stay within a half)
#pragma unroll
    for (int off = 1; off < 16; off <<= 1)
#pragma unroll
      for (int r = 0; r < 8; ++r)
        partial[r] += __shfl_xor(partial[r], off, 32);

    if (c < 8) {
      int e = base + 8 * half + c;
      float v = partial[c];
      a_out[e] = v;
      atomicMaxF(&amax[dst_idx[e]], v);
    }
  }
}

// ---------- kernel 3: ex = exp(a - amax[dst]); denom += ex ----------
__global__ void k_softmax(const int* __restrict__ dst_idx,
                          const float* __restrict__ amax,
                          float* __restrict__ a_ex, float* __restrict__ denom, int E) {
  int i = blockIdx.x * blockDim.x + threadIdx.x;
  if (i >= E) return;
  int d = dst_idx[i];
  float m = amax[d];
  if (!(m > -3.0e38f)) m = 0.0f;  // mirror reference isfinite() guard
  float ex = expf(a_ex[i] - m);
  a_ex[i] = ex;
  atomicAdd(&denom[d], ex);
}

// ---------- kernel 4: recompute h_ue, scatter agg[dst] += h * ex/denom ----------
__global__ void k_edge_pass3(const float* __restrict__ src_feats,
                             const float* __restrict__ edge_feats,
                             const int* __restrict__ src_idx,
                             const int* __restrict__ dst_idx,
                             const float* __restrict__ W_src,
                             const float* __restrict__ a_ex,
                             const float* __restrict__ denom,
                             float* __restrict__ agg, int nTiles) {
  int lane = threadIdx.x & 31;
  int half = lane >> 4, c = lane & 15;
  int wave = blockIdx.x * (blockDim.x >> 5) + wave_in_block();
  int nWaves = gridDim.x * (blockDim.x >> 5);

  v16bf B[4][3];
#pragma unroll
  for (int nt = 0; nt < 4; ++nt)
#pragma unroll
    for (int kt = 0; kt < 3; ++kt)
      B[nt][kt] = load_B_wT(W_src, 96, nt * 16 + c, kt, half);

  for (int t = wave; t < nTiles; t += nWaves) {
    int base = t * 16;
    v16bf A[3];
#pragma unroll
    for (int kt = 0; kt < 3; ++kt)
      A[kt] = load_A_edge(src_feats, edge_feats, src_idx, base + c, kt, half);

    v8f acc[4];
#pragma unroll
    for (int nt = 0; nt < 4; ++nt) {
      acc[nt] = (v8f){0.f, 0.f, 0.f, 0.f, 0.f, 0.f, 0.f, 0.f};
#pragma unroll
      for (int kt = 0; kt < 3; ++kt)
        acc[nt] = __builtin_amdgcn_wmma_f32_16x16x32_bf16(false, A[kt], false, B[nt][kt],
                                                          (short)0, acc[nt], false, false);
    }

#pragma unroll
    for (int r = 0; r < 8; ++r) {
      int e = base + r + 8 * half;
      int d = dst_idx[e];
      float sa = a_ex[e] / fmaxf(denom[d], 1e-9f);
      float* o = agg + d * 64 + c;
      atomicAdd(o +  0, acc[0][r] * sa);
      atomicAdd(o + 16, acc[1][r] * sa);
      atomicAdd(o + 32, acc[2][r] * sa);
      atomicAdd(o + 48, acc[3][r] * sa);
    }
  }
}

// ---------- host launcher ----------
extern "C" void kernel_launch(void* const* d_in, const int* in_sizes, int n_in,
                              void* d_out, int out_size, void* d_ws, size_t ws_size,
                              hipStream_t stream) {
  const float* src_feats  = (const float*)d_in[0];
  const float* edge_feats = (const float*)d_in[1];
  const float* dst_feats  = (const float*)d_in[2];
  const float* W_src      = (const float*)d_in[3];
  const float* W_dst      = (const float*)d_in[4];
  const int*   src_idx    = (const int*)d_in[5];
  const int*   dst_idx    = (const int*)d_in[6];
  float* agg = (float*)d_out;

  const int E = in_sizes[5];
  const int N = in_sizes[2] / 64;  // dst_feats rows (DST_DIM = 64)
  // Problem sizes are tile-divisible (E = 1.6M, N = 50000; both % 16 == 0).
  const int nTilesE = E / 16;
  const int nTilesN = N / 16;

  // workspace layout: dst_proj[N*64] | amax[N] | denom[N] | a_ex[E]   (~20 MB)
  float* dst_proj = (float*)d_ws;
  float* amax  = dst_proj + (size_t)N * 64;
  float* denom = amax + N;
  float* a_ex  = denom + N;

  dim3 blk(256);  // 8 wave32s

  k_init<<<dim3(2048), blk, 0, stream>>>(agg, amax, denom, N * 64, N);

  int dpBlocks = (nTilesN + 7) / 8;
  k_dstproj<<<dim3(dpBlocks), blk, 0, stream>>>(dst_feats, W_dst, dst_proj, nTilesN);

  k_edge_pass1<<<dim3(2048), blk, 0, stream>>>(src_feats, edge_feats, src_idx, dst_idx,
                                               W_src, dst_proj, a_ex, amax, nTilesE);

  k_softmax<<<dim3((E + 255) / 256), blk, 0, stream>>>(dst_idx, amax, a_ex, denom, E);

  k_edge_pass3<<<dim3(2048), blk, 0, stream>>>(src_feats, edge_feats, src_idx, dst_idx,
                                               W_src, a_ex, denom, agg, nTilesE);
}
